// ICLAttention_38328288149850
// MI455X (gfx1250) — compile-verified, matrix operand order checked by reading, not run on hardware
//
#include <hip/hip_runtime.h>
#include <hip/hip_bf16.h>

typedef __attribute__((ext_vector_type(16))) __bf16 v16bf;
typedef __attribute__((ext_vector_type(8)))  __bf16 v8bf;
typedef __attribute__((ext_vector_type(4)))  __bf16 v4bf;
typedef __attribute__((ext_vector_type(8)))  float  v8f;

static constexpr int   cB = 2, cS = 2048, cE = 1024, cH = 16, cDH = 64;
static constexpr int   cM = cB * cS;                   // 4096 token rows
static constexpr float cSC2 = 0.03125f * 1.4426950408889634f;  // (1/sqrt(E)) * log2(e)

// LDS byte offset of a generic pointer to __shared__ data: the flat LDS
// aperture maps addr[31:0] directly to the wave-relative LDS offset.
__device__ __forceinline__ unsigned lds_off(const void* p) {
  return (unsigned)(unsigned long long)p;
}

// One async DMA of 16B/lane (512B/wave) from global into LDS (ASYNCcnt path).
__device__ __forceinline__ void async_ldst16(unsigned lds_byte, const void* gaddr) {
  asm volatile("global_load_async_to_lds_b128 %0, %1, off"
               :: "v"(lds_byte), "v"(gaddr) : "memory");
}

// Load a 16-bit WMMA A/B fragment: two 16B chunks per lane
// (lanes 0-15: K {0..7,16..23}; lanes 16-31: K {8..15,24..31}; caller passes
// pointers already offset by khalf = (lane>>4)*8).
__device__ __forceinline__ v16bf ld_frag(const __bf16* c0, const __bf16* c1) {
  v8bf lo = *reinterpret_cast<const v8bf*>(c0);
  v8bf hi = *reinterpret_cast<const v8bf*>(c1);
  return __builtin_shufflevector(lo, hi, 0,1,2,3,4,5,6,7,8,9,10,11,12,13,14,15);
}

__device__ __forceinline__ v8f wmma_bf16(v16bf a, v16bf b, v8f c) {
  return __builtin_amdgcn_wmma_f32_16x16x32_bf16(false, a, false, b,
                                                 (short)0, c, false, false);
}

__device__ __forceinline__ float redmax16(float v) {
#pragma unroll
  for (int m = 8; m >= 1; m >>= 1) v = fmaxf(v, __shfl_xor(v, m, 32));
  return v;
}

__global__ void cvt_bf16x4_kernel(const float* __restrict__ in,
                                  __bf16* __restrict__ out, int n4) {
  int i = blockIdx.x * blockDim.x + threadIdx.x;
  if (i < n4) {
    const float4 f = reinterpret_cast<const float4*>(in)[i];
    v4bf o;
    o[0] = (__bf16)f.x; o[1] = (__bf16)f.y; o[2] = (__bf16)f.z; o[3] = (__bf16)f.w;
    reinterpret_cast<v4bf*>(out)[i] = o;
  }
}

// D[m,n] = sum_k A[m,k]*W[n,k].  All 8 waves of a block share one m-tile, so
// the contiguous 32KB A tile is staged once into LDS via async DMA; W rows
// stream from global (L2-resident) with prefetch.
// MODE 0: bf16 [B,H,S,DH] | MODE 1: bf16 [B,H,DH,S] | MODE 2: f32 [cM,cE]
template <int MODE>
__global__ __launch_bounds__(256, 1)
void gemm_bt_kernel(const __bf16* __restrict__ A,
                    const __bf16* __restrict__ W,
                    void* __restrict__ outp) {
  __shared__ __align__(16) __bf16 atile[16 * cE];   // 32 KB
  const int lane  = threadIdx.x & 31;
  const int wv    = threadIdx.x >> 5;
  const int wave  = blockIdx.x * 8 + wv;
  const int tn    = wave & (cE / 16 - 1);
  const int tm    = wave >> 6;                      // uniform across the block
  const int l15   = lane & 15;
  const int khalf = (lane >> 4) << 3;

  {  // cooperative async staging of the A tile (8 waves x 8 x 512B = 32KB)
    const char* gbase = (const char*)(A + (size_t)tm * 16 * cE);
    const unsigned lbase = lds_off(atile);
#pragma unroll
    for (int i = 0; i < 8; ++i) {
      const unsigned boff = (unsigned)(((wv * 8 + i) * 32 + lane) * 16);
      async_ldst16(lbase + boff, gbase + boff);
    }
    asm volatile("s_wait_asynccnt 0x0" ::: "memory");
    __syncthreads();
  }

  const __bf16* ap = atile + l15 * cE + khalf;                     // LDS
  const __bf16* bp = W + (size_t)(tn * 16 + l15) * cE + khalf;     // global
  v8f acc = {};
#pragma unroll 4
  for (int k = 0; k < cE; k += 32) {
    __builtin_prefetch(bp + 128, 0, 1);    // global_prefetch_b8, 256B ahead
    v16bf af = ld_frag(ap, ap + 16);       // ds_load_b128 x2
    v16bf bf = ld_frag(bp, bp + 16);
    acc = wmma_bf16(af, bf, acc);
    ap += 32; bp += 32;
  }

  const int rowoff = (lane >> 4) << 3;
#pragma unroll
  for (int r = 0; r < 8; ++r) {
    const int m = tm * 16 + rowoff + r;
    const int n = tn * 16 + l15;
    const float val = acc[r];
    if (MODE == 0) {
      const int b = m >> 11, s = m & (cS - 1);
      const int h = n >> 6,  d = n & (cDH - 1);
      reinterpret_cast<__bf16*>(outp)[(((size_t)(b * cH + h) * cS + s) * cDH) + d] = (__bf16)val;
    } else if (MODE == 1) {
      const int b = m >> 11, s = m & (cS - 1);
      const int h = n >> 6,  d = n & (cDH - 1);
      reinterpret_cast<__bf16*>(outp)[(((size_t)(b * cH + h) * cDH + d) * cS) + s] = (__bf16)val;
    } else {
      reinterpret_cast<float*>(outp)[(size_t)m * cE + n] = val;
    }
  }
}

// Flash attention: one wave per (b,h,16-query tile). K/V tiles double-buffered
// in LDS via async DMA; online softmax in exp2 domain; row-sums via one extra
// WMMA against an all-ones fragment instead of 32 shuffle-adds.
__global__ __launch_bounds__(128, 1)
void flash_attn_kernel(const __bf16* __restrict__ Qh,
                       const __bf16* __restrict__ Kh,
                       const __bf16* __restrict__ VhT,
                       __bf16* __restrict__ attn) {
  __shared__ __align__(16) __bf16 kls[4][2][32 * cDH];   // 2 x 4KB per wave
  __shared__ __align__(16) __bf16 vls[4][2][cDH * 32];   // 2 x 4KB per wave
  __shared__ __align__(16) __bf16 pls[4][16 * 32];       // 1KB per wave

  const int lane   = threadIdx.x & 31;
  const int wid    = threadIdx.x >> 5;
  const int gw     = blockIdx.x * 4 + wid;
  const int qt     = gw & 127;
  const int bh     = gw >> 7;
  const int l15    = lane & 15;
  const int khalf  = (lane >> 4) << 3;
  const int rowoff = (lane >> 4) << 3;

  const __bf16* Qp = Qh  + (size_t)bh * cS * cDH;
  const __bf16* Kp = Kh  + (size_t)bh * cS * cDH;
  const __bf16* Vp = VhT + (size_t)bh * cDH * cS;

  const __bf16* qrp = Qp + (size_t)(qt * 16 + l15) * cDH + khalf;
  const v16bf qf0 = ld_frag(qrp,      qrp + 16);
  const v16bf qf1 = ld_frag(qrp + 32, qrp + 48);

  v16bf onesf;
#pragma unroll
  for (int i = 0; i < 16; ++i) onesf[i] = (__bf16)1.0f;

  float mrow[8], lrow[8];
  v8f o0 = {}, o1 = {}, o2 = {}, o3 = {};
#pragma unroll
  for (int r = 0; r < 8; ++r) { mrow[r] = -3.0e38f; lrow[r] = 0.0f; }

  const int nsteps = (qt + 2) >> 1;   // trailing half-tile is fully masked

  // issue 16 async DMAs (8 for K-tile 32x64, 8 for V-tile 64x32) into buf
  auto issue_tiles = [&](int st, int buf) {
    const int tbase = st * 32;
    const unsigned kb = lds_off(&kls[wid][buf][0]);
    const unsigned vb = lds_off(&vls[wid][buf][0]);
#pragma unroll
    for (int i = 0; i < 8; ++i) {
      const int kr = i * 4 + (lane >> 3), kc = (lane & 7) * 8;
      async_ldst16(kb + (unsigned)(kr * cDH + kc) * 2,
                   Kp + (size_t)(tbase + kr) * cDH + kc);
      const int vr = i * 8 + (lane >> 2), vc = (lane & 3) * 8;
      async_ldst16(vb + (unsigned)(vr * 32 + vc) * 2,
                   Vp + (size_t)vr * cS + tbase + vc);
    }
  };

  issue_tiles(0, 0);
  __bf16* lp = &pls[wid][0];

  for (int st = 0; st < nsteps; ++st) {
    const int cur = st & 1;
    if (st + 1 < nsteps) {
      issue_tiles(st + 1, cur ^ 1);
      asm volatile("s_wait_asynccnt 0x10" ::: "memory");  // current tile done
    } else {
      asm volatile("s_wait_asynccnt 0x0" ::: "memory");
    }
    const __bf16* kt = &kls[wid][cur][0];
    const __bf16* vt = &vls[wid][cur][0];
    const int tbase = st * 32;

    // scores S = Q · K^T  (two 16-key halves, DH=64 => 2 k-steps each)
    const __bf16* kp0 = kt + l15 * cDH + khalf;
    const __bf16* kp1 = kp0 + 16 * cDH;
    v8f s0 = {}, s1 = {};
    s0 = wmma_bf16(qf0, ld_frag(kp0,      kp0 + 16), s0);
    s0 = wmma_bf16(qf1, ld_frag(kp0 + 32, kp0 + 48), s0);
    s1 = wmma_bf16(qf0, ld_frag(kp1,      kp1 + 16), s1);
    s1 = wmma_bf16(qf1, ld_frag(kp1 + 32, kp1 + 48), s1);

    // mask + scale (exp2 domain) + online max, stash probs to LDS
    const int t0 = tbase + l15, t1 = t0 + 16;
#pragma unroll
    for (int r = 0; r < 8; ++r) {
      const int  srow = qt * 16 + rowoff + r;
      const bool v0 = (t0 < srow) || (srow == 0 && t0 == 0);  // (0,0) exception
      const bool v1 = (t1 < srow);
      const float x0 = v0 ? s0[r] * cSC2 : -__builtin_inff();
      const float x1 = v1 ? s1[r] * cSC2 : -__builtin_inff();
      const float mnew = fmaxf(mrow[r], redmax16(fmaxf(x0, x1)));
      const float fsc  = exp2f(mrow[r] - mnew);
      const float p0   = exp2f(x0 - mnew);
      const float p1   = exp2f(x1 - mnew);
      mrow[r] = mnew;
      lrow[r] *= fsc;
      o0[r] *= fsc; o1[r] *= fsc; o2[r] *= fsc; o3[r] *= fsc;
      lp[(rowoff + r) * 32 + l15]      = (__bf16)p0;
      lp[(rowoff + r) * 32 + l15 + 16] = (__bf16)p1;
    }
    asm volatile("s_wait_dscnt 0x0" ::: "memory");

    // probs back as an A fragment (16x32); row-sums via P · ones in one WMMA
    const __bf16* pr = lp + l15 * 32 + khalf;
    const v16bf pf = ld_frag(pr, pr + 16);
    v8f lsum = {};
    lsum = wmma_bf16(pf, onesf, lsum);
#pragma unroll
    for (int r = 0; r < 8; ++r) lrow[r] += lsum[r];

    // O += P · V  (V staged transposed [DH x 32], B fragments load like A)
    const __bf16* vp = vt + l15 * 32 + khalf;
    o0 = wmma_bf16(pf, ld_frag(vp, vp + 16), o0);  vp += 16 * 32;
    o1 = wmma_bf16(pf, ld_frag(vp, vp + 16), o1);  vp += 16 * 32;
    o2 = wmma_bf16(pf, ld_frag(vp, vp + 16), o2);  vp += 16 * 32;
    o3 = wmma_bf16(pf, ld_frag(vp, vp + 16), o3);
  }

  // normalize, merge heads into [B,S,E] bf16
  const int b = bh >> 4, h = bh & 15;
#pragma unroll
  for (int r = 0; r < 8; ++r) {
    const float inv  = 1.0f / lrow[r];
    const int   srow = qt * 16 + rowoff + r;
    __bf16* op = attn + ((size_t)(b * cS + srow)) * cE + h * cDH + l15;
    op[0]  = (__bf16)(o0[r] * inv);
    op[16] = (__bf16)(o1[r] * inv);
    op[32] = (__bf16)(o2[r] * inv);
    op[48] = (__bf16)(o3[r] * inv);
  }
}

extern "C" void kernel_launch(void* const* d_in, const int* in_sizes, int n_in,
                              void* d_out, int out_size, void* d_ws, size_t ws_size,
                              hipStream_t stream) {
  (void)in_sizes; (void)n_in; (void)out_size; (void)ws_size;
  const float* q  = (const float*)d_in[0];
  const float* k  = (const float*)d_in[1];
  const float* v  = (const float*)d_in[2];
  const float* Wq = (const float*)d_in[3];
  const float* Wk = (const float*)d_in[4];
  const float* Wv = (const float*)d_in[5];
  const float* Wo = (const float*)d_in[6];
  float* out = (float*)d_out;
  char*  ws  = (char*)d_ws;

  const size_t MB = 1ull << 20;
  __bf16* qb   = (__bf16*)(ws +  0 * MB);  // 8 MB; reused as attn buffer later
  __bf16* kb   = (__bf16*)(ws +  8 * MB);
  __bf16* vb   = (__bf16*)(ws + 16 * MB);
  __bf16* wqb  = (__bf16*)(ws + 24 * MB);
  __bf16* wkb  = (__bf16*)(ws + 26 * MB);
  __bf16* wvb  = (__bf16*)(ws + 28 * MB);
  __bf16* wob  = (__bf16*)(ws + 30 * MB);
  __bf16* Qh   = (__bf16*)(ws + 32 * MB);  // [B,H,S,DH]
  __bf16* Kh   = (__bf16*)(ws + 40 * MB);  // [B,H,S,DH]
  __bf16* VhT  = (__bf16*)(ws + 48 * MB);  // [B,H,DH,S]
  __bf16* attn = qb;                       // q-bf16 dead after Q projection

  const int nAct = cM * cE;
  const int nW   = cE * cE;
  cvt_bf16x4_kernel<<<(nAct / 4 + 255) / 256, 256, 0, stream>>>(q,  qb,  nAct / 4);
  cvt_bf16x4_kernel<<<(nAct / 4 + 255) / 256, 256, 0, stream>>>(k,  kb,  nAct / 4);
  cvt_bf16x4_kernel<<<(nAct / 4 + 255) / 256, 256, 0, stream>>>(v,  vb,  nAct / 4);
  cvt_bf16x4_kernel<<<(nW   / 4 + 255) / 256, 256, 0, stream>>>(Wq, wqb, nW / 4);
  cvt_bf16x4_kernel<<<(nW   / 4 + 255) / 256, 256, 0, stream>>>(Wk, wkb, nW / 4);
  cvt_bf16x4_kernel<<<(nW   / 4 + 255) / 256, 256, 0, stream>>>(Wv, wvb, nW / 4);
  cvt_bf16x4_kernel<<<(nW   / 4 + 255) / 256, 256, 0, stream>>>(Wo, wob, nW / 4);

  const int gemmBlocks = (cM / 16) * (cE / 16) / 8;   // 2048
  gemm_bt_kernel<0><<<gemmBlocks, 256, 0, stream>>>(qb, wqb, Qh);
  gemm_bt_kernel<0><<<gemmBlocks, 256, 0, stream>>>(kb, wkb, Kh);
  gemm_bt_kernel<1><<<gemmBlocks, 256, 0, stream>>>(vb, wvb, VhT);

  flash_attn_kernel<<<(cB * cH * (cS / 16)) / 4, 128, 0, stream>>>(Qh, Kh, VhT, attn);

  gemm_bt_kernel<2><<<gemmBlocks, 256, 0, stream>>>(attn, wob, out);
}
